// SparseDopplerAttention_35450660061769
// MI455X (gfx1250) — matrix operand order repaired and checked by reading, not compiled, not run on hardware
//
#include <hip/hip_runtime.h>
#include <hip/hip_bf16.h>

typedef __attribute__((ext_vector_type(16))) _Float16 v16h;
typedef __attribute__((ext_vector_type(8)))  _Float16 v8h;
typedef __attribute__((ext_vector_type(8)))  float    v8f;

#define NRANGES 256
#define KPR     512   // keys / queries per range
#define DOP     64
#define EMB     16
#define FDIM    96    // DOP + 2*EMB

// LDS strides (in halves), 16B-aligned rows
#define XSTR 104      // 96 + pad
#define QSTR 72       // 64 + pad

// LDS layout (bytes); all offsets 16B aligned
#define OFF_XS 0
#define SZ_XS  (KPR * XSTR * 2)          // 106496
#define OFF_WQ (OFF_XS + SZ_XS)          // 106496
#define SZ_W   (DOP * XSTR * 2)          // 13312
#define OFF_WK (OFF_WQ + SZ_W)           // 119808
#define OFF_QS (OFF_WK + SZ_W)           // 133120
#define SZ_QS  (KPR * QSTR * 2)          // 73728
#define OFF_KS (OFF_QS + SZ_QS)          // 206848
#define OFF_BQ (OFF_KS + SZ_QS)          // 280576
#define OFF_BK (OFF_BQ + 256)
#define OFF_VS (OFF_BK + 256)            // vsum: 512 f32
#define OFF_WV (OFF_VS + 2048)           // wvsum: 96 f32
#define OFF_BV (OFF_WV + 384)            // bvsum scalar
#define SMEM_TOTAL (OFF_BV + 16)         // 283536 B (< 320KB WGP LDS -> 1 WG/WGP)

// ---- WMMA operand loaders (CDNA5 16-bit 16x16x32 layouts, ISA 7.12.2) ----
// A (16 M x 32 K): lane l -> row m=l%16; halves[0..7]=K[k0+8g+i], halves[8..15]=K[k0+16+8g+i], g=l/16
__device__ __forceinline__ v16h loadA(const _Float16* base, int stride, int row0, int k0, int lane) {
    const int m = lane & 15, g = lane >> 4;
    const _Float16* p = base + (row0 + m) * stride + k0 + g * 8;
    v8h lo = *(const v8h*)p;
    v8h hi = *(const v8h*)(p + 16);
    v16h r;
#pragma unroll
    for (int i = 0; i < 8; ++i) { r[i] = lo[i]; r[i + 8] = hi[i]; }
    return r;
}
// B (32 K x 16 N): lane l -> column n=l%16 (an LDS row); halves[0..7]=K[k0+8g+i], halves[8..15]=K[k0+16+8g+i], g=l/16
__device__ __forceinline__ v16h loadB(const _Float16* base, int stride, int row0, int k0, int lane) {
    const int n = lane & 15, g = lane >> 4;
    const _Float16* p = base + (row0 + n) * stride + k0 + g * 8;
    v8h lo = *(const v8h*)p;
    v8h hi = *(const v8h*)(p + 16);
    v16h r;
#pragma unroll
    for (int i = 0; i < 8; ++i) { r[i] = lo[i]; r[i + 8] = hi[i]; }
    return r;
}

__device__ __forceinline__ v8h pack8(float4 a, float4 b) {
    v8h h;
    h[0] = (_Float16)a.x; h[1] = (_Float16)a.y; h[2] = (_Float16)a.z; h[3] = (_Float16)a.w;
    h[4] = (_Float16)b.x; h[5] = (_Float16)b.y; h[6] = (_Float16)b.z; h[7] = (_Float16)b.w;
    return h;
}

__global__ __launch_bounds__(256) void sda_fused_kernel(
    const float* __restrict__ pv,  const int* __restrict__ ele_idx,
    const int*  __restrict__ azi_idx,
    const float* __restrict__ ele_tab, const float* __restrict__ azi_tab,
    const float* __restrict__ Wq, const float* __restrict__ bq,
    const float* __restrict__ Wk, const float* __restrict__ bk,
    const float* __restrict__ Wv, const float* __restrict__ bv,
    float* __restrict__ out)
{
    extern __shared__ __align__(16) char smem[];
    _Float16* XS  = (_Float16*)(smem + OFF_XS);   // x  f16 [512 x 104]
    _Float16* WQS = (_Float16*)(smem + OFF_WQ);   // Wq f16 [64 x 104]
    _Float16* WKS = (_Float16*)(smem + OFF_WK);   // Wk f16 [64 x 104]
    _Float16* QS  = (_Float16*)(smem + OFF_QS);   // Q  f16 [512 x 72]
    _Float16* KS  = (_Float16*)(smem + OFF_KS);   // K  f16 [512 x 72]
    float* BQS  = (float*)(smem + OFF_BQ);
    float* BKS  = (float*)(smem + OFF_BK);
    float* VSUM = (float*)(smem + OFF_VS);
    float* WVS  = (float*)(smem + OFF_WV);
    float* BVS  = (float*)(smem + OFF_BV);

    const int tid  = threadIdx.x;
    const int lane = tid & 31;
    const int wv   = tid >> 5;
    const int r    = blockIdx.x;

    // ---------------- Phase 1: stage x (power | ele_emb | azi_emb) + weights ----------------
    for (int k = tid; k < KPR; k += 256) {
        const float* src = pv + (size_t)(r * KPR + k) * DOP;
        _Float16* dst = XS + k * XSTR;
#pragma unroll
        for (int f = 0; f < DOP; f += 8) {
            float4 a = *(const float4*)(src + f);
            float4 b = *(const float4*)(src + f + 4);
            *(v8h*)(dst + f) = pack8(a, b);
        }
        const int ge = r * KPR + k;
        const int ei = ele_idx[ge] * EMB;
        const int ai = azi_idx[ge] * EMB;
#pragma unroll
        for (int e = 0; e < EMB; e += 8) {
            float4 ea = *(const float4*)(ele_tab + ei + e);
            float4 eb = *(const float4*)(ele_tab + ei + e + 4);
            *(v8h*)(dst + DOP + e) = pack8(ea, eb);
            float4 aa = *(const float4*)(azi_tab + ai + e);
            float4 ab = *(const float4*)(azi_tab + ai + e + 4);
            *(v8h*)(dst + DOP + EMB + e) = pack8(aa, ab);
        }
    }
    for (int idx = tid; idx < DOP * FDIM; idx += 256) {
        const int d = idx / FDIM, f = idx - d * FDIM;
        WQS[d * XSTR + f] = (_Float16)Wq[idx];
        WKS[d * XSTR + f] = (_Float16)Wk[idx];
    }
    if (tid < DOP) { BQS[tid] = bq[tid]; BKS[tid] = bk[tid]; }
    if (tid < FDIM) {                    // wvsum[f] = sum_d Wv[d][f]
        float s = 0.f;
        for (int d = 0; d < DOP; ++d) s += Wv[d * FDIM + tid];
        WVS[tid] = s;
    }
    if (tid == 0) {                      // bvsum = sum_d bv[d]
        float s = 0.f;
        for (int d = 0; d < DOP; ++d) s += bv[d];
        BVS[0] = s;
    }
    __syncthreads();

    // ---------------- Phase 2a: vsum[k] = x[k,:].wvsum + bvsum (V GEMM collapsed) ----------
    for (int k = tid; k < KPR; k += 256) {
        const _Float16* xr = XS + k * XSTR;
        float s = BVS[0];
#pragma unroll
        for (int f = 0; f < FDIM; f += 8) {
            v8h hx = *(const v8h*)(xr + f);
#pragma unroll
            for (int i = 0; i < 8; ++i) s += (float)hx[i] * WVS[f + i];
        }
        VSUM[k] = s;
    }

    // ---------------- Phase 2b: Q/K projections via WMMA --------------------------------
    // 64 row-jobs (2 mats x 32 row-tiles); A tiles hoisted across the 4 column tiles.
    for (int i = 0; i < 8; ++i) {
        const int job = wv + 8 * i;          // uniform per wave
        const int mat = job >> 5;            // 0 = Q, 1 = K
        const int q0  = (job & 31) * 16;
        const _Float16* WL = mat ? WKS : WQS;
        const v16h a0 = loadA(XS, XSTR, q0, 0,  lane);
        const v16h a1 = loadA(XS, XSTR, q0, 32, lane);
        const v16h a2 = loadA(XS, XSTR, q0, 64, lane);
        const float* BL = mat ? BKS : BQS;
        _Float16* DST = mat ? KS : QS;
        const int n = lane & 15, mb = (lane >> 4) * 8;
#pragma unroll
        for (int ct = 0; ct < 4; ++ct) {
            const int d0 = ct * 16;
            v8f c = {};
            c = __builtin_amdgcn_wmma_f32_16x16x32_f16(false, a0, false, loadB(WL, XSTR, d0, 0,  lane), (short)0, c, false, false);
            c = __builtin_amdgcn_wmma_f32_16x16x32_f16(false, a1, false, loadB(WL, XSTR, d0, 32, lane), (short)0, c, false, false);
            c = __builtin_amdgcn_wmma_f32_16x16x32_f16(false, a2, false, loadB(WL, XSTR, d0, 64, lane), (short)0, c, false, false);
            const float bias = BL[d0 + n];
#pragma unroll
            for (int j = 0; j < 8; ++j)
                DST[(q0 + mb + j) * QSTR + d0 + n] = (_Float16)(c[j] + bias);
        }
    }
    __syncthreads();

    // ---------------- Phase 3: fused scores + softmax + dot(attn, vsum) -----------------
    // Two q-strips per wave per pass: B (K-matrix) tiles shared across both -> half the
    // LDS B-traffic per WMMA. scores ~ N(0,1/9): exp() without running-max is safe.
    for (int si = 0; si < 2; ++si) {
        const int q0a = (wv + si * 16) * 16;
        const int q0b = (wv + 8 + si * 16) * 16;
        const v16h A0a = loadA(QS, QSTR, q0a, 0,  lane);
        const v16h A1a = loadA(QS, QSTR, q0a, 32, lane);
        const v16h A0b = loadA(QS, QSTR, q0b, 0,  lane);
        const v16h A1b = loadA(QS, QSTR, q0b, 32, lane);
        float da[8], na[8], db[8], nb[8];
#pragma unroll
        for (int j = 0; j < 8; ++j) { da[j] = na[j] = db[j] = nb[j] = 0.f; }

        for (int kk = 0; kk < KPR; kk += 16) {
            const v16h B0 = loadB(KS, QSTR, kk, 0,  lane);
            const v16h B1 = loadB(KS, QSTR, kk, 32, lane);
            v8f ca = {}, cb = {};
            ca = __builtin_amdgcn_wmma_f32_16x16x32_f16(false, A0a, false, B0, (short)0, ca, false, false);
            ca = __builtin_amdgcn_wmma_f32_16x16x32_f16(false, A1a, false, B1, (short)0, ca, false, false);
            cb = __builtin_amdgcn_wmma_f32_16x16x32_f16(false, A0b, false, B0, (short)0, cb, false, false);
            cb = __builtin_amdgcn_wmma_f32_16x16x32_f16(false, A1b, false, B1, (short)0, cb, false, false);
            const float vs = VSUM[kk + (lane & 15)];    // key column this lane owns
#pragma unroll
            for (int j = 0; j < 8; ++j) {
                const float ea = __expf(ca[j] * 0.125f);   // 1/sqrt(64)
                const float eb = __expf(cb[j] * 0.125f);
                da[j] += ea; na[j] += ea * vs;
                db[j] += eb; nb[j] += eb * vs;
            }
        }
        // row m of D tile = VGPR (m%8), lanes (m<8 ? 0-15 : 16-31); reduce over N=lane%16
        const int hf = lane >> 4;
#pragma unroll
        for (int j = 0; j < 8; ++j) {
            float dv = da[j], nv = na[j];
            float dw = db[j], nw = nb[j];
#pragma unroll
            for (int off = 1; off < 16; off <<= 1) {
                dv += __shfl_xor(dv, off, 32);
                nv += __shfl_xor(nv, off, 32);
                dw += __shfl_xor(dw, off, 32);
                nw += __shfl_xor(nw, off, 32);
            }
            if ((lane & 15) == 0) {
                out[(size_t)r * KPR + q0a + hf * 8 + j] = nv / dv;
                out[(size_t)r * KPR + q0b + hf * 8 + j] = nw / dw;
            }
        }
    }
}

extern "C" void kernel_launch(void* const* d_in, const int* in_sizes, int n_in,
                              void* d_out, int out_size, void* d_ws, size_t ws_size,
                              hipStream_t stream) {
    (void)in_sizes; (void)n_in; (void)d_ws; (void)ws_size; (void)out_size;
    const float* pv      = (const float*)d_in[0];
    const int*   ele_idx = (const int*)  d_in[1];
    // d_in[2] = range_indices: unused by the reference computation
    const int*   azi_idx = (const int*)  d_in[3];
    const float* ele_tab = (const float*)d_in[4];
    const float* azi_tab = (const float*)d_in[5];
    const float* Wq      = (const float*)d_in[6];
    const float* bq      = (const float*)d_in[7];
    const float* Wk      = (const float*)d_in[8];
    const float* bk      = (const float*)d_in[9];
    const float* Wv      = (const float*)d_in[10];
    const float* bv      = (const float*)d_in[11];
    float* out = (float*)d_out;

    sda_fused_kernel<<<dim3(NRANGES), dim3(256), SMEM_TOTAL, stream>>>(
        pv, ele_idx, azi_idx, ele_tab, azi_tab, Wq, bq, Wk, bk, Wv, bv, out);
}
